// IBSConnectivityMatrixGenerator_50869592655003
// MI455X (gfx1250) — matrix operand order repaired and checked by reading, not compiled, not run on hardware
//
#include <hip/hip_runtime.h>
#include <math.h>

// ---------------------------------------------------------------------------
// IBS connectivity matrices on gfx1250 (CDNA5).
// Inputs:  eeg1, eeg2: (4, 64, 2048) f32.  Output: (4, 6, 7, 64, 64) f32.
// All bilinear stages (DFT fwd/inv, PLV, power-corr, time-corr) run on
// v_wmma_f32_16x16x32_f16 with a 16x64 output tile per wave (4 independent
// accumulator chains). Sign/abs pairwise stats + coherence are VALU
// reduction kernels.
// ---------------------------------------------------------------------------

typedef __attribute__((ext_vector_type(16))) _Float16 v16h;
typedef __attribute__((ext_vector_type(8)))  float    v8f;

#define TLEN 2048
#define NSIG 512          // 2 inputs * 4 batches * 64 channels
#define NBIN 384          // spectrum bins kept (covers k in [4,360], padded)
#define NCH  64
#define NB   4
#define FEATSTRIDE 4096   // 64*64
#define OUTBATCH   (6 * 7 * FEATSTRIDE)

union V16HU { v16h v; uint4 u[2]; };

// A-matrix 16x32 f16 tile per ISA layout:
//   lane<16 : M=lane,   K = {k0+0..7, k0+16..23}
//   lane>=16: M=lane-16,K = {k0+8..15, k0+24..31}
__device__ __forceinline__ v16h load_a16x32(const _Float16* __restrict__ A,
                                            int lda, int m0, int k0, int lane) {
  const _Float16* row = A + (size_t)(m0 + (lane & 15)) * (size_t)lda;
  int kb = k0 + ((lane & 16) ? 8 : 0);
  V16HU r;
  r.u[0] = *reinterpret_cast<const uint4*>(row + kb);
  r.u[1] = *reinterpret_cast<const uint4*>(row + kb + 16);
  __builtin_prefetch(row + kb + 64, 0, 1);
  return r.v;
}

// B-matrix 32x16 tile; we compute C = X * Y^T, so B[k][n] = Y[n][k]:
//   lane<16 : N=lane,   K = {k0+0..7, k0+16..23}
//   lane>=16: N=lane-16,K = {k0+8..15, k0+24..31}
__device__ __forceinline__ v16h load_b32x16(const _Float16* __restrict__ Y,
                                            int ldy, int n0, int k0, int lane) {
  const _Float16* row = Y + (size_t)(n0 + (lane & 15)) * (size_t)ldy;
  int kb = k0 + ((lane & 16) ? 8 : 0);
  V16HU r;
  r.u[0] = *reinterpret_cast<const uint4*>(row + kb);
  r.u[1] = *reinterpret_cast<const uint4*>(row + kb + 16);
  __builtin_prefetch(row + kb + 64, 0, 1);
  return r.v;
}

// Generic batched C[m][n] (+)= alpha * sum_k A[m][k] * B[n][k].
// One wave (32 lanes) per 16x64 output tile: A tile shared across 4 N-subtiles,
// 4 independent WMMA accumulator chains. grid = (N/64, M/16, batches).
__global__ __launch_bounds__(32)
void wmma_gemm_xyT(const _Float16* __restrict__ A, const _Float16* __restrict__ B,
                   float* __restrict__ C,
                   int K, int lda, int ldb, int ldc,
                   long long sA, long long sB, long long sC,
                   float alpha, int accumulate) {
  int lane = threadIdx.x;
  int n0 = blockIdx.x * 64;
  int m0 = blockIdx.y * 16;
  A += (long long)blockIdx.z * sA;
  B += (long long)blockIdx.z * sB;
  C += (long long)blockIdx.z * sC;

  v8f acc0 = {}, acc1 = {}, acc2 = {}, acc3 = {};
  for (int k0 = 0; k0 < K; k0 += 32) {
    v16h a  = load_a16x32(A, lda, m0, k0, lane);
    v16h b0 = load_b32x16(B, ldb, n0,      k0, lane);
    v16h b1 = load_b32x16(B, ldb, n0 + 16, k0, lane);
    v16h b2 = load_b32x16(B, ldb, n0 + 32, k0, lane);
    v16h b3 = load_b32x16(B, ldb, n0 + 48, k0, lane);
    acc0 = __builtin_amdgcn_wmma_f32_16x16x32_f16(false, a, false, b0, (short)0, acc0, false, false);
    acc1 = __builtin_amdgcn_wmma_f32_16x16x32_f16(false, a, false, b1, (short)0, acc1, false, false);
    acc2 = __builtin_amdgcn_wmma_f32_16x16x32_f16(false, a, false, b2, (short)0, acc2, false, false);
    acc3 = __builtin_amdgcn_wmma_f32_16x16x32_f16(false, a, false, b3, (short)0, acc3, false, false);
  }

  // C/D layout: VGPR r, lane L -> (M = m0 + r + 8*(L>=16), N = n0 + L%16)
  int mbase = m0 + ((lane & 16) ? 8 : 0);
  int cbase = n0 + (lane & 15);
  v8f accs[4] = {acc0, acc1, acc2, acc3};
#pragma unroll
  for (int u = 0; u < 4; ++u) {
#pragma unroll
    for (int r = 0; r < 8; ++r) {
      float v = alpha * accs[u][r];
      float* dst = C + (size_t)(mbase + r) * (size_t)ldc + cbase + 16 * u;
      if (accumulate) v += *dst;
      *dst = v;
    }
  }
}

// ---------------------------------------------------------------------------
// DFT basis tables (f16), both [k][t] (fwd B operand) and [t][k] (inv B operand)
__global__ void build_basis(_Float16* __restrict__ Ckt, _Float16* __restrict__ Skt,
                            _Float16* __restrict__ Ctk, _Float16* __restrict__ Stk) {
  int t = blockIdx.x * blockDim.x + threadIdx.x;   // 0..2047
  int k = blockIdx.y;                              // 0..383
  int ph = (k * t) & (TLEN - 1);                   // exact phase mod T
  float ang = (float)ph * (6.283185307179586f / (float)TLEN);
  float c = __cosf(ang), s = __sinf(ang);
  Ckt[(size_t)k * TLEN + t] = (_Float16)c;
  Skt[(size_t)k * TLEN + t] = (_Float16)s;
  Ctk[(size_t)t * NBIN + k] = (_Float16)c;
  Stk[(size_t)t * NBIN + k] = (_Float16)s;
}

__global__ void to_half(const float* __restrict__ e1, const float* __restrict__ e2,
                        _Float16* __restrict__ Xh) {
  size_t idx = (size_t)blockIdx.x * blockDim.x + threadIdx.x;  // < 512*2048
  const size_t half = (size_t)(NSIG / 2) * TLEN;
  float v = (idx < half) ? e1[idx] : e2[idx - half];
  Xh[idx] = (_Float16)v;
}

// Band mask: spectrum -> masked f16 copies + per-band power spectrum (coherence)
__global__ void band_mask(const float* __restrict__ Fre, const float* __restrict__ Fim,
                          _Float16* __restrict__ FreBh, _Float16* __restrict__ FimBh,
                          float* __restrict__ Pband, float lo, float hi) {
  int idx = blockIdx.x * blockDim.x + threadIdx.x;  // < 512*384
  int k = idx % NBIN;
  float freq = (float)k * 0.125f;                   // k * FS / T = k/8
  float m = (freq >= lo && freq <= hi) ? 1.0f : 0.0f;
  float re = Fre[idx] * m, im = Fim[idx] * m;
  FreBh[idx] = (_Float16)re;
  FimBh[idx] = (_Float16)im;
  Pband[idx] = re * re + im * im;
}

// Analytic signal -> f (over ReA), p (over ImA), phase, cos/sin (f16)
__global__ void analytic_ew(float* __restrict__ ReA, float* __restrict__ ImA,
                            float* __restrict__ Ph,
                            _Float16* __restrict__ CosH, _Float16* __restrict__ SinH) {
  size_t idx = (size_t)blockIdx.x * blockDim.x + threadIdx.x;  // < 512*2048
  const float sc = 2.0f / (float)TLEN;
  float re = ReA[idx] * sc;
  float im = ImA[idx] * sc;
  float mag = sqrtf(re * re + im * im) + 1e-20f;
  CosH[idx] = (_Float16)(re / mag);
  SinH[idx] = (_Float16)(im / mag);
  Ph[idx] = atan2f(im, re);
  ReA[idx] = re;            // bandpassed signal f
  ImA[idx] = re * re;       // power p = f^2
}

// Per-signal stats: mean/1/(std+eps) of f and p (ddof=1), and sum(p)
__global__ __launch_bounds__(256)
void stats_kernel(const float* __restrict__ f, const float* __restrict__ p,
                  float* __restrict__ stats) {
  int s = blockIdx.x, tid = threadIdx.x;
  const float* fr = f + (size_t)s * TLEN;
  const float* pr = p + (size_t)s * TLEN;
  float sf = 0.f, sf2 = 0.f, sp = 0.f, sp2 = 0.f;
  for (int t = tid; t < TLEN; t += 256) {
    float a = fr[t]; sf += a; sf2 += a * a;
    float q = pr[t]; sp += q; sp2 += q * q;
  }
  __shared__ float r0[256], r1[256], r2[256], r3[256];
  r0[tid] = sf; r1[tid] = sf2; r2[tid] = sp; r3[tid] = sp2;
  __syncthreads();
  for (int off = 128; off > 0; off >>= 1) {
    if (tid < off) {
      r0[tid] += r0[tid + off]; r1[tid] += r1[tid + off];
      r2[tid] += r2[tid + off]; r3[tid] += r3[tid + off];
    }
    __syncthreads();
  }
  if (tid == 0) {
    const float T = (float)TLEN;
    float fm = r0[0] / T;
    float fv = fmaxf((r1[0] - T * fm * fm) / (T - 1.0f), 0.0f);
    float pm = r2[0] / T;
    float pv = fmaxf((r3[0] - T * pm * pm) / (T - 1.0f), 0.0f);
    stats[s * 8 + 0] = fm;
    stats[s * 8 + 1] = 1.0f / (sqrtf(fv) + 1e-8f);
    stats[s * 8 + 2] = pm;
    stats[s * 8 + 3] = 1.0f / (sqrtf(pv) + 1e-8f);
    stats[s * 8 + 4] = r2[0];               // sum(p) for wPLI denom
  }
}

__global__ void zscore_write(const float* __restrict__ f, const float* __restrict__ p,
                             const float* __restrict__ stats,
                             _Float16* __restrict__ FzH, _Float16* __restrict__ PzH) {
  size_t idx = (size_t)blockIdx.x * blockDim.x + threadIdx.x;  // < 512*2048
  int s = (int)(idx >> 11);
  FzH[idx] = (_Float16)((f[idx] - stats[s * 8 + 0]) * stats[s * 8 + 1]);
  PzH[idx] = (_Float16)((p[idx] - stats[s * 8 + 2]) * stats[s * 8 + 3]);
}

// PLV magnitude -> feature 0
__global__ void plv_finalize(const float* __restrict__ Re, const float* __restrict__ Im,
                             float* __restrict__ out, int band) {
  int idx = blockIdx.x * blockDim.x + threadIdx.x;  // < 4*4096
  int b = idx >> 12, rem = idx & 4095;
  float re = Re[idx], im = Im[idx];
  out[((size_t)(b * 6 + band) * 7 + 0) * FEATSTRIDE + rem] =
      sqrtf(re * re + im * im) * (1.0f / (float)TLEN);
}

// PLI / wPLI / phase-diff: block per (b, i); loops j, strided over t.
__global__ __launch_bounds__(256)
void pairwise_pli(const float* __restrict__ Ph, const float* __restrict__ P,
                  const float* __restrict__ stats, float* __restrict__ out, int band) {
  int i = blockIdx.x, b = blockIdx.y, tid = threadIdx.x;
  __shared__ float sph[TLEN];
  __shared__ float spw[TLEN];
  __shared__ float ra[256], rb[256], rc[256];

  int s1 = b * NCH + i;
  for (int t = tid; t < TLEN; t += 256) {
    sph[t] = Ph[(size_t)s1 * TLEN + t];
    spw[t] = P[(size_t)s1 * TLEN + t];
  }
  __syncthreads();
  float S1 = stats[s1 * 8 + 4];

  for (int j = 0; j < NCH; ++j) {
    int s2 = 256 + b * NCH + j;
    const float* ph2 = Ph + (size_t)s2 * TLEN;
    const float* p2  = P  + (size_t)s2 * TLEN;
    float as = 0.f, aw = 0.f, ad = 0.f;
    for (int t = tid; t < TLEN; t += 256) {
      float pd = sph[t] - ph2[t];
      float sg = (pd > 0.f) ? 1.0f : ((pd < 0.f) ? -1.0f : 0.0f);
      as += sg;
      aw += sg * 0.5f * (spw[t] + p2[t]);
      ad += fabsf(pd);
    }
    ra[tid] = as; rb[tid] = aw; rc[tid] = ad;
    __syncthreads();
    for (int off = 128; off > 0; off >>= 1) {
      if (tid < off) {
        ra[tid] += ra[tid + off];
        rb[tid] += rb[tid + off];
        rc[tid] += rc[tid + off];
      }
      __syncthreads();
    }
    if (tid == 0) {
      float S2 = stats[s2 * 8 + 4];
      size_t base = (size_t)(b * 6 + band) * 7 * FEATSTRIDE + (size_t)i * NCH + j;
      out[base + 1 * FEATSTRIDE] = fabsf(ra[0]) * (1.0f / (float)TLEN);
      out[base + 2 * FEATSTRIDE] = fabsf(rb[0]) / (0.5f * (S1 + S2) + 1e-8f);
      out[base + 5 * FEATSTRIDE] = rc[0] * (1.0f / (float)TLEN);
    }
    __syncthreads();
  }
}

// Coherence over rfft bins (mean over F = T/2+1 = 1025 bins; out-of-band -> 0)
__global__ __launch_bounds__(256)
void pairwise_coh(const float* __restrict__ Pband, float* __restrict__ out, int band) {
  int i = blockIdx.x, b = blockIdx.y, tid = threadIdx.x;
  __shared__ float sp1[NBIN];
  __shared__ float red[256];
  int s1 = b * NCH + i;
  for (int k = tid; k < NBIN; k += 256) sp1[k] = Pband[(size_t)s1 * NBIN + k];
  __syncthreads();
  for (int j = 0; j < NCH; ++j) {
    const float* p2 = Pband + (size_t)(256 + b * NCH + j) * NBIN;
    float acc = 0.f;
    for (int k = tid; k < NBIN; k += 256) {
      float pw = sp1[k] * p2[k];
      acc += pw / (pw + 1e-8f);
    }
    red[tid] = acc;
    __syncthreads();
    for (int off = 128; off > 0; off >>= 1) {
      if (tid < off) red[tid] += red[tid + off];
      __syncthreads();
    }
    if (tid == 0) {
      out[((size_t)(b * 6 + band) * 7 + 3) * FEATSTRIDE + (size_t)i * NCH + j] =
          red[0] * (1.0f / 1025.0f);
    }
    __syncthreads();
  }
}

// ---------------------------------------------------------------------------
extern "C" void kernel_launch(void* const* d_in, const int* in_sizes, int n_in,
                              void* d_out, int out_size, void* d_ws, size_t ws_size,
                              hipStream_t stream) {
  (void)in_sizes; (void)n_in; (void)out_size; (void)ws_size;
  const float* eeg1 = (const float*)d_in[0];
  const float* eeg2 = (const float*)d_in[1];
  float* out = (float*)d_out;

  // ---- workspace carve-up (aligned 256B) ----
  size_t off = 0;
  auto carve = [&](size_t bytes) {
    void* p = (char*)d_ws + off;
    off += (bytes + 255) & ~(size_t)255;
    return p;
  };
  _Float16* Wcos_kt = (_Float16*)carve((size_t)NBIN * TLEN * 2);
  _Float16* Wsin_kt = (_Float16*)carve((size_t)NBIN * TLEN * 2);
  _Float16* Wcos_tk = (_Float16*)carve((size_t)TLEN * NBIN * 2);
  _Float16* Wsin_tk = (_Float16*)carve((size_t)TLEN * NBIN * 2);
  _Float16* Xh      = (_Float16*)carve((size_t)NSIG * TLEN * 2);
  float*    Fre     = (float*)carve((size_t)NSIG * NBIN * 4);
  float*    Fim     = (float*)carve((size_t)NSIG * NBIN * 4);
  _Float16* FreBh   = (_Float16*)carve((size_t)NSIG * NBIN * 2);
  _Float16* FimBh   = (_Float16*)carve((size_t)NSIG * NBIN * 2);
  float*    Pband   = (float*)carve((size_t)NSIG * NBIN * 4);
  float*    ReA     = (float*)carve((size_t)NSIG * TLEN * 4);  // -> f
  float*    ImA     = (float*)carve((size_t)NSIG * TLEN * 4);  // -> p
  float*    Ph      = (float*)carve((size_t)NSIG * TLEN * 4);
  _Float16* CosH    = (_Float16*)carve((size_t)NSIG * TLEN * 2);
  _Float16* SinH    = (_Float16*)carve((size_t)NSIG * TLEN * 2);
  _Float16* FzH     = (_Float16*)carve((size_t)NSIG * TLEN * 2);
  _Float16* PzH     = (_Float16*)carve((size_t)NSIG * TLEN * 2);
  float*    Stats   = (float*)carve((size_t)NSIG * 8 * 4);
  float*    RePLV   = (float*)carve((size_t)NB * FEATSTRIDE * 4);
  float*    ImPLV   = (float*)carve((size_t)NB * FEATSTRIDE * 4);

  // ---- stage 0: basis + input conversion ----
  build_basis<<<dim3(TLEN / 256, NBIN), 256, 0, stream>>>(Wcos_kt, Wsin_kt, Wcos_tk, Wsin_tk);
  to_half<<<(NSIG * TLEN) / 256, 256, 0, stream>>>(eeg1, eeg2, Xh);

  // ---- stage 1: forward DFT (bins 0..383): F[s][k] = sum_t x cos  /  -sum_t x sin
  wmma_gemm_xyT<<<dim3(NBIN / 64, NSIG / 16, 1), 32, 0, stream>>>(
      Xh, Wcos_kt, Fre, TLEN, TLEN, TLEN, NBIN, 0, 0, 0, 1.0f, 0);
  wmma_gemm_xyT<<<dim3(NBIN / 64, NSIG / 16, 1), 32, 0, stream>>>(
      Xh, Wsin_kt, Fim, TLEN, TLEN, TLEN, NBIN, 0, 0, 0, -1.0f, 0);

  static const float LO[6] = {0.5f, 0.5f, 4.0f, 8.0f, 13.0f, 30.0f};
  static const float HI[6] = {45.0f, 4.0f, 8.0f, 13.0f, 30.0f, 45.0f};

  const long long sAB = (long long)NCH * TLEN;     // batch stride in signals

  for (int band = 0; band < 6; ++band) {
    band_mask<<<(NSIG * NBIN) / 256, 256, 0, stream>>>(
        Fre, Fim, FreBh, FimBh, Pband, LO[band], HI[band]);

    // inverse DFT -> analytic: ReA = FreB*cos - FimB*sin ; ImA = FreB*sin + FimB*cos
    wmma_gemm_xyT<<<dim3(TLEN / 64, NSIG / 16, 1), 32, 0, stream>>>(
        FreBh, Wcos_tk, ReA, NBIN, NBIN, NBIN, TLEN, 0, 0, 0, 1.0f, 0);
    wmma_gemm_xyT<<<dim3(TLEN / 64, NSIG / 16, 1), 32, 0, stream>>>(
        FimBh, Wsin_tk, ReA, NBIN, NBIN, NBIN, TLEN, 0, 0, 0, -1.0f, 1);
    wmma_gemm_xyT<<<dim3(TLEN / 64, NSIG / 16, 1), 32, 0, stream>>>(
        FreBh, Wsin_tk, ImA, NBIN, NBIN, NBIN, TLEN, 0, 0, 0, 1.0f, 0);
    wmma_gemm_xyT<<<dim3(TLEN / 64, NSIG / 16, 1), 32, 0, stream>>>(
        FimBh, Wcos_tk, ImA, NBIN, NBIN, NBIN, TLEN, 0, 0, 0, 1.0f, 1);

    analytic_ew<<<(NSIG * TLEN) / 256, 256, 0, stream>>>(ReA, ImA, Ph, CosH, SinH);
    stats_kernel<<<NSIG, 256, 0, stream>>>(ReA, ImA, Stats);
    zscore_write<<<(NSIG * TLEN) / 256, 256, 0, stream>>>(ReA, ImA, Stats, FzH, PzH);

    // ---- batched 64x64 feature GEMMs (grid.z = batch) ----
    const _Float16* Cos2 = CosH + (size_t)256 * TLEN;
    const _Float16* Sin2 = SinH + (size_t)256 * TLEN;
    const _Float16* Pz2  = PzH  + (size_t)256 * TLEN;
    const _Float16* Fz2  = FzH  + (size_t)256 * TLEN;
    dim3 g44(NCH / 64, NCH / 16, NB);   // (1, 4, 4)

    // Re PLV = cos1*cos2^T + sin1*sin2^T ; Im PLV = sin1*cos2^T - cos1*sin2^T
    wmma_gemm_xyT<<<g44, 32, 0, stream>>>(CosH, Cos2, RePLV, TLEN, TLEN, TLEN, NCH,
                                          sAB, sAB, FEATSTRIDE, 1.0f, 0);
    wmma_gemm_xyT<<<g44, 32, 0, stream>>>(SinH, Sin2, RePLV, TLEN, TLEN, TLEN, NCH,
                                          sAB, sAB, FEATSTRIDE, 1.0f, 1);
    wmma_gemm_xyT<<<g44, 32, 0, stream>>>(SinH, Cos2, ImPLV, TLEN, TLEN, TLEN, NCH,
                                          sAB, sAB, FEATSTRIDE, 1.0f, 0);
    wmma_gemm_xyT<<<g44, 32, 0, stream>>>(CosH, Sin2, ImPLV, TLEN, TLEN, TLEN, NCH,
                                          sAB, sAB, FEATSTRIDE, -1.0f, 1);

    // power-corr (feat 4) and time-corr (feat 6) straight into d_out
    wmma_gemm_xyT<<<g44, 32, 0, stream>>>(PzH, Pz2, out + (size_t)(band * 7 + 4) * FEATSTRIDE,
                                          TLEN, TLEN, TLEN, NCH,
                                          sAB, sAB, (long long)OUTBATCH,
                                          1.0f / (float)TLEN, 0);
    wmma_gemm_xyT<<<g44, 32, 0, stream>>>(FzH, Fz2, out + (size_t)(band * 7 + 6) * FEATSTRIDE,
                                          TLEN, TLEN, TLEN, NCH,
                                          sAB, sAB, (long long)OUTBATCH,
                                          1.0f / (float)TLEN, 0);

    plv_finalize<<<(NB * FEATSTRIDE) / 256, 256, 0, stream>>>(RePLV, ImPLV, out, band);
    pairwise_pli<<<dim3(NCH, NB), 256, 0, stream>>>(Ph, ImA, Stats, out, band);
    pairwise_coh<<<dim3(NCH, NB), 256, 0, stream>>>(Pband, out, band);
  }
}
